// MinibatchDiscrimination_22728966930708
// MI455X (gfx1250) — matrix operand order repaired and checked by reading, not compiled, not run on hardware
//
#include <hip/hip_runtime.h>

typedef _Float16 v16h __attribute__((ext_vector_type(16)));
typedef _Float16 v8h  __attribute__((ext_vector_type(8)));
typedef float    v8f  __attribute__((ext_vector_type(8)));

#define IN_F   1024
#define OUT_F  1280
#define BEX    256     // extra output features
#define NROW   256     // batch
#define JDIM   4096    // BEX * K(=16), GEMM N dimension

// ---------------------------------------------------------------------------
// Kernel 1: convert x (f32, 256x1024) -> f16, contiguous.
// ---------------------------------------------------------------------------
__global__ void mbd_cvt_x(const float* __restrict__ x, _Float16* __restrict__ x16) {
  int i = blockIdx.x * blockDim.x + threadIdx.x;
  x16[i] = (_Float16)x[i];
}

// ---------------------------------------------------------------------------
// Kernel 2: transpose + convert T (f32, 1024 x 4096) -> Wt (f16, 4096 x 1024).
// LDS-tiled so both the read and the write are coalesced.
// ---------------------------------------------------------------------------
__global__ void mbd_cvt_T(const float* __restrict__ T, _Float16* __restrict__ Wt) {
  __shared__ float tile[32][33];
  const int tx = threadIdx.x, ty = threadIdx.y;
  const int j0 = blockIdx.x * 32;   // column block in T  (0..4095)
  const int i0 = blockIdx.y * 32;   // row block in T     (0..1023)
  tile[ty][tx] = T[(i0 + ty) * JDIM + (j0 + tx)];
  __syncthreads();
  Wt[(j0 + ty) * IN_F + (i0 + tx)] = (_Float16)tile[tx][ty];
}

// ---------------------------------------------------------------------------
// Kernel 3: M = x16 (256x1024) @ W (1024x4096) via v_wmma_f32_16x16x32_f16.
// One wave owns one 16x16 tile of M; 32 K-steps of 32.
// B is consumed from Wt (row = output column, 1024 contiguous K halves) so
// each lane's 16-half B slice is contiguous (two b128 loads).
// ---------------------------------------------------------------------------
__global__ void __launch_bounds__(256)
mbd_gemm_wmma(const _Float16* __restrict__ A,   // x16: 256 x 1024
              const _Float16* __restrict__ Bt,  // Wt : 4096 x 1024
              float* __restrict__ M)            // out: 256 x 4096 (f32)
{
  const int lane = threadIdx.x & 31;
  const int wave = threadIdx.x >> 5;
  const int tile = blockIdx.x * 8 + wave;  // 4096 tiles total
  const int tm   = tile & 15;              // 16 tiles along rows (256/16)
  const int tn   = tile >> 4;              // 256 tiles along cols (4096/16)
  const int rb   = tm * 16;
  const int cb   = tn * 16;
  const int r    = lane & 15;              // A row / B column within tile
  const int sel  = lane >> 4;              // half-wave selector

  // A lane slice: halves 0..15 = K (sel*16 + 0..15)
  const _Float16* arow = A  + (rb + r) * IN_F + sel * 16;
  // B lane slice: halves 0..15 = K (sel*16 + 0..15) at column cb+r
  const _Float16* brow = Bt + (cb + r) * IN_F + sel * 16;

  v8f acc = {};
#pragma unroll 4
  for (int k0 = 0; k0 < IN_F; k0 += 32) {
    v8h alo = *(const v8h*)(arow + k0);
    v8h ahi = *(const v8h*)(arow + k0 + 8);
    v8h blo = *(const v8h*)(brow + k0);
    v8h bhi = *(const v8h*)(brow + k0 + 8);
    v16h a, b;
#pragma unroll
    for (int j = 0; j < 8; ++j) {
      a[j] = alo[j]; a[j + 8] = ahi[j];
      b[j] = blo[j]; b[j + 8] = bhi[j];
    }
    // 8 args: (neg_a, A, neg_b, B, c_mod, C, reuse_a, reuse_b)
    acc = __builtin_amdgcn_wmma_f32_16x16x32_f16(
        false, a, false, b, (short)0, acc, false, false);
  }

  // D layout: VGPR v -> row (v + 8*sel), column = lane%16
  float* dst = M + (rb + sel * 8) * JDIM + (cb + r);
#pragma unroll
  for (int v = 0; v < 8; ++v) dst[v * JDIM] = acc[v];
}

// ---------------------------------------------------------------------------
// Kernel 4: copy x into out[:, 0:1024].
// ---------------------------------------------------------------------------
__global__ void mbd_copy_x(const float* __restrict__ x, float* __restrict__ out) {
  int i = blockIdx.x * blockDim.x + threadIdx.x;
  int n = i >> 10;          // /1024
  int c = i & 1023;
  out[n * OUT_F + c] = x[i];
}

// ---------------------------------------------------------------------------
// Kernel 5: for each feature b, out[n,1024+b] = sum_m exp(-||M[n,b,:]-M[m,b,:]||_1) - 1
// One block per b; 256 threads = one per n; all 256 K=16 rows staged in LDS
// (16 KB).  Per m-step every lane reads the same LDS word -> broadcast.
// ---------------------------------------------------------------------------
__global__ void __launch_bounds__(256)
mbd_pairwise(const float* __restrict__ M, float* __restrict__ out) {
  __shared__ float sm[NROW * 16];
  const int b = blockIdx.x;
  const int n = threadIdx.x;

  const float4* src = (const float4*)(M + n * JDIM + b * 16);
  float4 r0 = src[0], r1 = src[1], r2 = src[2], r3 = src[3];
  float4* dst4 = (float4*)(sm + n * 16);
  dst4[0] = r0; dst4[1] = r1; dst4[2] = r2; dst4[3] = r3;
  __syncthreads();

  float mn[16];
  mn[0]=r0.x; mn[1]=r0.y; mn[2]=r0.z; mn[3]=r0.w;
  mn[4]=r1.x; mn[5]=r1.y; mn[6]=r1.z; mn[7]=r1.w;
  mn[8]=r2.x; mn[9]=r2.y; mn[10]=r2.z; mn[11]=r2.w;
  mn[12]=r3.x; mn[13]=r3.y; mn[14]=r3.z; mn[15]=r3.w;

  float acc = 0.0f;
  for (int m = 0; m < NROW; ++m) {
    const float* row = sm + m * 16;
    float l1 = 0.0f;
#pragma unroll
    for (int k = 0; k < 16; ++k) l1 += fabsf(mn[k] - row[k]);
    acc += __expf(-l1);          // m==n contributes exp(0)=1
  }
  out[n * OUT_F + IN_F + b] = acc - 1.0f;   // subtract self term
}

// ---------------------------------------------------------------------------
extern "C" void kernel_launch(void* const* d_in, const int* in_sizes, int n_in,
                              void* d_out, int out_size, void* d_ws, size_t ws_size,
                              hipStream_t stream) {
  const float* x = (const float*)d_in[0];   // 256 x 1024
  const float* T = (const float*)d_in[1];   // 1024 x 256 x 16  (= 1024 x 4096)
  float* out     = (float*)d_out;           // 256 x 1280

  char* ws = (char*)d_ws;
  _Float16* x16 = (_Float16*)ws;                                 // 512 KB
  _Float16* Wt  = (_Float16*)(ws + (size_t)512 * 1024);          // 8 MB
  float*    M   = (float*)(ws + (size_t)512 * 1024 + (size_t)8 * 1024 * 1024); // 4 MB

  mbd_cvt_x<<<(NROW * IN_F) / 256, 256, 0, stream>>>(x, x16);
  mbd_cvt_T<<<dim3(JDIM / 32, IN_F / 32), dim3(32, 32), 0, stream>>>(T, Wt);
  mbd_gemm_wmma<<<512, 256, 0, stream>>>(x16, Wt, M);
  mbd_copy_x<<<(NROW * IN_F) / 256, 256, 0, stream>>>(x, out);
  mbd_pairwise<<<BEX, NROW, 0, stream>>>(M, out);
}
